// Attention_37641093382387
// MI455X (gfx1250) — matrix operand order repaired and checked by reading, not compiled, not run on hardware
//
#include <hip/hip_runtime.h>
#include <hip/hip_bf16.h>

// ---------------------------------------------------------------------------
// CDNA5 (gfx1250) attention pipeline.
// - all contractions on v_wmma_f32_16x16x32_bf16 (fp32 accumulate)
// - every operand tile is a row-major 2D tile staged global->LDS by the
//   Tensor Data Mover (tensor_load_to_lds), double-buffered and software-
//   pipelined on TENSORcnt (in-order per-wave completion => wait cnt<=1).
// D_MODEL=2048, H=16, hd=128, B=2, T=2048.
// ---------------------------------------------------------------------------

typedef __bf16 bf16;
typedef __attribute__((ext_vector_type(16))) __bf16 v16bf;
typedef __attribute__((ext_vector_type(8)))  __bf16 v8bf;
typedef __attribute__((ext_vector_type(8)))  float  v8f;
typedef __attribute__((ext_vector_type(4)))  unsigned int u32x4;
typedef __attribute__((ext_vector_type(4)))  int i32x4;
typedef __attribute__((ext_vector_type(8)))  int i32x8;

#define D_MODEL 2048
#define NHEADS  16
#define HDIM    128
#define BATCH   2
#define SEQ     2048

static __device__ __forceinline__ v8f wmma_bf16(v16bf a, v16bf b, v8f c) {
  return __builtin_amdgcn_wmma_f32_16x16x32_bf16(false, a, false, b, (short)0, c,
                                                 false, false);
}
static __device__ __forceinline__ v16bf cat16(v8bf lo, v8bf hi) {
  return __builtin_shufflevector(lo, hi, 0, 1, 2, 3, 4, 5, 6, 7,
                                 8, 9, 10, 11, 12, 13, 14, 15);
}

// ---- TDM: async copy of a 2D bf16 tile (tile_w x tile_h) global -> LDS ----
// D# group0/group1 packed per CDNA5 ISA 8.3/8.4. pad fields add 16B to each
// LDS row (pad_amt=3 -> 4 DWORDs) every 2^(ival+3) bytes.
static __device__ __forceinline__ void tdm_load_2d(
    unsigned lds_addr, const void* gsrc, unsigned tile_w, unsigned tile_h,
    unsigned stride_elems, unsigned pad_ival, unsigned pad_amt) {
  unsigned long long ga = (unsigned long long)(uintptr_t)gsrc;
  u32x4 g0;
  g0[0] = 1u;                                        // count=1 (user D#)
  g0[1] = lds_addr;                                  // LDS byte offset
  g0[2] = (unsigned)ga;                              // global_addr[31:0]
  g0[3] = ((unsigned)(ga >> 32) & 0x01FFFFFFu) | (2u << 30);  // [56:32] | type=2
  i32x8 g1;
  g1[0] = (int)((1u << 16) |                         // data_size = 2 bytes
                (1u << 20) |                         // pad_enable
                (pad_ival << 22) | (pad_amt << 25));
  g1[1] = (int)((stride_elems & 0xFFFFu) << 16);     // tensor_dim0[15:0]
  g1[2] = (int)((stride_elems >> 16) | (0xFFFFu << 16));  // dim0 hi | dim1 lo
  g1[3] = (int)(tile_w << 16);                       // tile_dim0
  g1[4] = (int)(tile_h & 0xFFFFu);                   // tile_dim1
  g1[5] = (int)stride_elems;                         // tensor_dim0_stride lo32
  g1[6] = 0;
  g1[7] = 0;
  i32x4 z4;
  z4[0] = z4[1] = z4[2] = z4[3] = 0;
  i32x8 z8;
  z8[0] = z8[1] = z8[2] = z8[3] = z8[4] = z8[5] = z8[6] = z8[7] = 0;
  __builtin_amdgcn_tensor_load_to_lds(g0, g1, z4, z4, z8, 0);
}

// ------------------------------- fp32 -> bf16 ------------------------------
__global__ void cvt_f32_bf16(const float* __restrict__ in, bf16* __restrict__ out,
                             size_t n8) {
  size_t i = (size_t)blockIdx.x * blockDim.x + threadIdx.x;
  if (i >= n8) return;
  const float4 a = ((const float4*)in)[2 * i];
  const float4 b = ((const float4*)in)[2 * i + 1];
  v8bf o;
  o[0] = (bf16)a.x; o[1] = (bf16)a.y; o[2] = (bf16)a.z; o[3] = (bf16)a.w;
  o[4] = (bf16)b.x; o[5] = (bf16)b.y; o[6] = (bf16)b.z; o[7] = (bf16)b.w;
  ((v8bf*)out)[i] = o;
}

// ---------------- fp32 [R,C] -> bf16 [C,R] (weights, one-time) -------------
__global__ __launch_bounds__(256) void cvt_transpose_f32_bf16(
    const float* __restrict__ in, bf16* __restrict__ out, int R, int C) {
  __shared__ float tile[32][33];
  const int tx = threadIdx.x & 31, ty = threadIdx.x >> 5;  // 32x8
  const int r0 = blockIdx.y * 32, c0 = blockIdx.x * 32;
  for (int i = 0; i < 4; ++i)
    tile[ty + i * 8][tx] = in[(size_t)(r0 + ty + i * 8) * C + c0 + tx];
  __syncthreads();
  for (int i = 0; i < 4; ++i)
    out[(size_t)(c0 + ty + i * 8) * R + r0 + tx] = (bf16)tile[tx][ty + i * 8];
}

// ------------------------------- tiled GEMM --------------------------------
// C[M,N](fp32,+bias) = A[M,K](bf16,row) @ Bt[N,K](bf16,row = B transposed)
// block tile 128x128, K-step 64, 256 threads = 8 waves (4 M x 2 N).
// A/Bt tiles (128 x 64) double-buffered in LDS, TDM-staged + pipelined.
__global__ __launch_bounds__(256) void gemm_bf16_f32(
    const bf16* __restrict__ A, const bf16* __restrict__ Bt,
    float* __restrict__ C, const float* __restrict__ bias,
    int M, int N, int K, int guard) {
  __shared__ bf16 As[2][128][72];    // [buf][row][k]: 128B rows + 16B TDM pad
  __shared__ bf16 Bst[2][128][72];   // [buf][col][k]

  const int tid  = threadIdx.x;
  const int wid  = tid >> 5;
  const int lane = tid & 31;
  const int half = lane >> 4;
  const int l16  = lane & 15;

  if (guard) { As[0][0][0] = (bf16)0.f; Bst[0][0][0] = (bf16)0.f; }

  const int mblk = blockIdx.y * 128;
  const int nblk = blockIdx.x * 128;
  const int wm = (wid & 3) * 32;   // wave row offset (2 m-tiles)
  const int wn = (wid >> 2) * 64;  // wave col offset (4 n-tiles)

  const bf16* abase = A + (size_t)mblk * K;
  const bf16* bbase = Bt + (size_t)nblk * K;

  v8f acc[2][4];
  for (int i = 0; i < 2; ++i)
    for (int j = 0; j < 4; ++j)
      for (int r = 0; r < 8; ++r) acc[i][j][r] = 0.0f;

  const int ntiles = K >> 6;  // K-steps of 64

  // prologue: stage tile 0
  if (wid == 0)
    tdm_load_2d((unsigned)(uintptr_t)&As[0][0][0], abase, 64, 128, (unsigned)K, 4, 3);
  else if (wid == 1)
    tdm_load_2d((unsigned)(uintptr_t)&Bst[0][0][0], bbase, 64, 128, (unsigned)K, 4, 3);

  for (int t = 0; t < ntiles; ++t) {
    const int buf = t & 1;
    // issue next tile into other buffer, then wait for current tile
    if (wid == 0) {
      if (t + 1 < ntiles) {
        tdm_load_2d((unsigned)(uintptr_t)&As[buf ^ 1][0][0],
                    abase + (size_t)(t + 1) * 64, 64, 128, (unsigned)K, 4, 3);
        __builtin_amdgcn_s_wait_tensorcnt((short)1);
      } else {
        __builtin_amdgcn_s_wait_tensorcnt((short)0);
      }
    } else if (wid == 1) {
      if (t + 1 < ntiles) {
        tdm_load_2d((unsigned)(uintptr_t)&Bst[buf ^ 1][0][0],
                    bbase + (size_t)(t + 1) * 64, 64, 128, (unsigned)K, 4, 3);
        __builtin_amdgcn_s_wait_tensorcnt((short)1);
      } else {
        __builtin_amdgcn_s_wait_tensorcnt((short)0);
      }
    }
    __syncthreads();

    // ---- 2 k-chunks of 32: fragments are two b128 LDS loads each ----
    for (int c = 0; c < 2; ++c) {
      v16bf afrag[2];
      for (int i = 0; i < 2; ++i) {
        int row = wm + i * 16 + l16;
        afrag[i] = cat16(*(const v8bf*)&As[buf][row][c * 32 + half * 8],
                         *(const v8bf*)&As[buf][row][c * 32 + 16 + half * 8]);
      }
      for (int j = 0; j < 4; ++j) {   // single live B-fragment
        int col = wn + j * 16 + l16;
        v16bf bfrag = cat16(*(const v8bf*)&Bst[buf][col][c * 32 + half * 8],
                            *(const v8bf*)&Bst[buf][col][c * 32 + 16 + half * 8]);
        acc[0][j] = wmma_bf16(afrag[0], bfrag, acc[0][j]);
        acc[1][j] = wmma_bf16(afrag[1], bfrag, acc[1][j]);
      }
    }
    __syncthreads();  // protect buf^1 before next issue overwrites it
  }

  // C layout: VGPR r = rows r / 8+r, lane%16 = col
  for (int i = 0; i < 2; ++i) {
    int row0 = mblk + wm + i * 16 + half * 8;
    for (int j = 0; j < 4; ++j) {
      int col = nblk + wn + j * 16 + l16;
      float bv = bias ? bias[col] : 0.0f;
      for (int r = 0; r < 8; ++r)
        C[(size_t)(row0 + r) * N + col] = acc[i][j][r] + bv;
    }
  }
}

// ------------------------------- RoPE + split ------------------------------
// qkv fp32 [B,T,3D] -> q,k roped bf16 [B,H,T,hd]; v -> bf16 [B,H,hd,T] (Vt)
__global__ void rope_split(const float* __restrict__ qkv, bf16* __restrict__ Qo,
                           bf16* __restrict__ Ko, bf16* __restrict__ Vt) {
  size_t gid = (size_t)blockIdx.x * blockDim.x + threadIdx.x;
  const size_t total = (size_t)BATCH * NHEADS * SEQ * 64;
  if (gid >= total) return;
  int d = (int)(gid & 63);
  size_t rest = gid >> 6;
  int t = (int)(rest % SEQ); rest /= SEQ;
  int h = (int)(rest % NHEADS);
  int b = (int)(rest / NHEADS);
  const int bh = b * NHEADS + h;

  const size_t rowoff = ((size_t)b * SEQ + t) * (3 * D_MODEL);
  const int c0 = h * HDIM + d, c1 = c0 + 64;
  float q0 = qkv[rowoff + c0],               q1 = qkv[rowoff + c1];
  float k0 = qkv[rowoff + D_MODEL + c0],     k1 = qkv[rowoff + D_MODEL + c1];
  float v0 = qkv[rowoff + 2 * D_MODEL + c0], v1 = qkv[rowoff + 2 * D_MODEL + c1];

  float freq = __powf(10000.0f, -(float)d * (1.0f / 64.0f));
  float th = (float)t * freq, cs, sn;
  __sincosf(th, &sn, &cs);

  size_t ob = (((size_t)bh) * SEQ + t) * HDIM + d;
  Qo[ob]      = (bf16)(q0 * cs - q1 * sn);
  Qo[ob + 64] = (bf16)(q1 * cs + q0 * sn);
  Ko[ob]      = (bf16)(k0 * cs - k1 * sn);
  Ko[ob + 64] = (bf16)(k1 * cs + k0 * sn);
  size_t vb0 = (((size_t)bh) * HDIM + d) * SEQ + t;        // Vt[bh][d][t]
  Vt[vb0]             = (bf16)v0;
  Vt[vb0 + 64 * SEQ]  = (bf16)v1;
}

// --------------------------- flash attention core --------------------------
// grid: (T/64, B*H). 128 threads = 4 waves, each wave owns 16 query rows.
// 32-key chunks; K + Vt tiles double-buffered in LDS via pipelined TDM.
__global__ __launch_bounds__(128) void flash_attn(
    const bf16* __restrict__ Qm, const bf16* __restrict__ Km,
    const bf16* __restrict__ Vt, bf16* __restrict__ Out, int guard) {
  __shared__ bf16 Ks[2][32][136];   // [buf][key][dim] (256B rows + 16B pad)
  __shared__ bf16 Vst[2][128][40];  // [buf][dim][key] (64B rows + 16B pad)
  __shared__ bf16 Ps[4][16][40];    // per-wave P (16x32), 16B-aligned rows

  const int tid  = threadIdx.x;
  const int wid  = tid >> 5;
  const int lane = tid & 31;
  const int half = lane >> 4;
  const int l16  = lane & 15;

  if (guard) { Ks[0][0][0] = (bf16)0.f; Vst[0][0][0] = (bf16)0.f; }

  const int bh = blockIdx.y;
  const int b = bh >> 4, h = bh & 15;
  const bf16* qp  = Qm + (size_t)bh * SEQ * HDIM;
  const bf16* kp  = Km + (size_t)bh * SEQ * HDIM;
  const bf16* vtp = Vt + (size_t)bh * HDIM * SEQ;

  const int qbase = blockIdx.x * 64 + wid * 16;
  const int nch   = 2 * blockIdx.x + 2;  // 32-key chunks within causal bound

  // Q fragments for the 4 contraction chunks of hd=128 (b128 global loads)
  v16bf qf[4];
  {
    const v8bf* q8 = (const v8bf*)(qp + (size_t)(qbase + l16) * HDIM);
    for (int c = 0; c < 4; ++c)
      qf[c] = cat16(q8[c * 4 + half], q8[c * 4 + half + 2]);
  }

  float mrow[8], lrow[8];
  v8f oacc[8];
  for (int r = 0; r < 8; ++r) { mrow[r] = -1e30f; lrow[r] = 0.0f; }
  for (int j = 0; j < 8; ++j)
    for (int r = 0; r < 8; ++r) oacc[j][r] = 0.0f;

  const float scale = 0.08838834764831845f;  // 1/sqrt(128)

  // prologue: stage chunk 0
  if (wid == 0)
    tdm_load_2d((unsigned)(uintptr_t)&Ks[0][0][0], kp, 128, 32, HDIM, 5, 3);
  else if (wid == 1)
    tdm_load_2d((unsigned)(uintptr_t)&Vst[0][0][0], vtp, 32, 128, SEQ, 3, 3);

  for (int ci = 0; ci < nch; ++ci) {
    const int kb = ci * 32;
    const int buf = ci & 1;
    if (wid == 0) {
      if (ci + 1 < nch) {
        tdm_load_2d((unsigned)(uintptr_t)&Ks[buf ^ 1][0][0],
                    kp + (size_t)(kb + 32) * HDIM, 128, 32, HDIM, 5, 3);
        __builtin_amdgcn_s_wait_tensorcnt((short)1);
      } else {
        __builtin_amdgcn_s_wait_tensorcnt((short)0);
      }
    } else if (wid == 1) {
      if (ci + 1 < nch) {
        tdm_load_2d((unsigned)(uintptr_t)&Vst[buf ^ 1][0][0],
                    vtp + kb + 32, 32, 128, SEQ, 3, 3);
        __builtin_amdgcn_s_wait_tensorcnt((short)1);
      } else {
        __builtin_amdgcn_s_wait_tensorcnt((short)0);
      }
    }
    __syncthreads();

    // ---- S = Q @ K^T (two 16x16 tiles, contraction hd=128 in 4 steps) ----
    v8f s[2];
    for (int t = 0; t < 2; ++t) {
      for (int r = 0; r < 8; ++r) s[t][r] = 0.0f;
      int n = t * 16 + l16;  // key index within chunk
      for (int c = 0; c < 4; ++c) {
        v16bf bf = cat16(*(const v8bf*)&Ks[buf][n][c * 32 + half * 8],
                         *(const v8bf*)&Ks[buf][n][c * 32 + 16 + half * 8]);
        s[t] = wmma_bf16(qf[c], bf, s[t]);
      }
    }

    // ---- scale + causal mask + streaming softmax ----
    for (int r = 0; r < 8; ++r) {
      int row = qbase + half * 8 + r;
      float p0 = (kb + l16 <= row) ? s[0][r] * scale : -1e30f;
      float p1 = (kb + 16 + l16 <= row) ? s[1][r] * scale : -1e30f;
      float x = fmaxf(p0, p1);
      for (int off = 1; off < 16; off <<= 1) x = fmaxf(x, __shfl_xor(x, off, 32));
      float mn = fmaxf(mrow[r], x);
      float alpha = __expf(mrow[r] - mn);
      mrow[r] = mn;
      float e0 = __expf(p0 - mn);
      float e1 = __expf(p1 - mn);
      float rs = e0 + e1;
      for (int off = 1; off < 16; off <<= 1) rs += __shfl_xor(rs, off, 32);
      lrow[r] = lrow[r] * alpha + rs;
      for (int j = 0; j < 8; ++j) oacc[j][r] *= alpha;
      Ps[wid][half * 8 + r][l16]      = (bf16)e0;   // P in C-layout
      Ps[wid][half * 8 + r][16 + l16] = (bf16)e1;
    }

    // reload P as A-fragment; same-wave DS ops are in-order
    v16bf pa = cat16(*(const v8bf*)&Ps[wid][l16][half * 8],
                     *(const v8bf*)&Ps[wid][l16][16 + half * 8]);

    // ---- O += P @ V (8 d-tiles, contraction = 32 keys) ----
    for (int j = 0; j < 8; ++j) {
      int n = j * 16 + l16;  // hd dim
      v16bf vbf = cat16(*(const v8bf*)&Vst[buf][n][half * 8],
                        *(const v8bf*)&Vst[buf][n][16 + half * 8]);
      oacc[j] = wmma_bf16(pa, vbf, oacc[j]);
    }
    __syncthreads();  // protect buf^1 before next issue overwrites it
  }

  // ---- finalize: O /= l, write bf16 [B,T, h*hd + d] ----
  for (int r = 0; r < 8; ++r) {
    float inv = 1.0f / lrow[r];
    int row = qbase + half * 8 + r;
    size_t base = ((size_t)b * SEQ + row) * D_MODEL + h * HDIM;
    for (int j = 0; j < 8; ++j)
      Out[base + j * 16 + l16] = (bf16)(oacc[j][r] * inv);
  }
}

// ------------------------------- launcher ----------------------------------
extern "C" void kernel_launch(void* const* d_in, const int* in_sizes, int n_in,
                              void* d_out, int out_size, void* d_ws, size_t ws_size,
                              hipStream_t stream) {
  const float* x     = (const float*)d_in[0];  // [B,T,D]
  const float* Wqkv  = (const float*)d_in[1];  // [D,3D]
  const float* Wproj = (const float*)d_in[2];  // [D,D]
  const float* bproj = (const float*)d_in[3];  // [D]
  float* out = (float*)d_out;                   // [B,T,D]

  const size_t BT = (size_t)BATCH * SEQ;               // 4096
  const size_t n_x = BT * D_MODEL;
  const size_t n_wqkv = (size_t)D_MODEL * 3 * D_MODEL;
  const size_t n_wproj = (size_t)D_MODEL * D_MODEL;
  const size_t n_qkv = BT * 3 * D_MODEL;
  const size_t n_head = BT * D_MODEL;

  char* ws = (char*)d_ws;
  size_t off = 0;
  auto alloc = [&](size_t bytes) {
    char* p = ws + off;
    off = (off + bytes + 255) & ~(size_t)255;
    return p;
  };
  bf16*  xb    = (bf16*)alloc(n_x * 2);
  bf16*  wqbT  = (bf16*)alloc(n_wqkv * 2);   // [3D, D]
  bf16*  wpbT  = (bf16*)alloc(n_wproj * 2);  // [D, D] transposed
  float* qkv   = (float*)alloc(n_qkv * 4);
  bf16*  qh    = (bf16*)alloc(n_head * 2);
  bf16*  kh    = (bf16*)alloc(n_head * 2);
  bf16*  vt    = (bf16*)alloc(n_head * 2);   // [B,H,hd,T]
  bf16*  attn  = (bf16*)alloc(n_head * 2);
  (void)ws_size;

  // 1) convert x; convert+transpose weights (one-time)
  cvt_f32_bf16<<<(unsigned)((n_x / 8 + 255) / 256), 256, 0, stream>>>(x, xb, n_x / 8);
  cvt_transpose_f32_bf16<<<dim3(3 * D_MODEL / 32, D_MODEL / 32), 256, 0, stream>>>(
      Wqkv, wqbT, D_MODEL, 3 * D_MODEL);
  cvt_transpose_f32_bf16<<<dim3(D_MODEL / 32, D_MODEL / 32), 256, 0, stream>>>(
      Wproj, wpbT, D_MODEL, D_MODEL);

  // 2) qkv = x @ Wqkv  (M=4096, N=6144, K=2048)
  gemm_bf16_f32<<<dim3(3 * D_MODEL / 128, (unsigned)(BT / 128)), 256, 0, stream>>>(
      xb, wqbT, qkv, nullptr, (int)BT, 3 * D_MODEL, D_MODEL, 0);

  // 3) RoPE + split (q,k per-head; v transposed per-head)
  {
    size_t total = (size_t)BATCH * NHEADS * SEQ * 64;
    rope_split<<<(unsigned)((total + 255) / 256), 256, 0, stream>>>(qkv, qh, kh, vt);
  }

  // 4) causal flash attention -> attn bf16 [B,T,D] (head-major in D)
  flash_attn<<<dim3(SEQ / 64, BATCH * NHEADS), 128, 0, stream>>>(qh, kh, vt, attn, 0);

  // 5) out = attn @ Wproj + bproj  (M=4096, N=2048, K=2048)
  gemm_bf16_f32<<<dim3(D_MODEL / 128, (unsigned)(BT / 128)), 256, 0, stream>>>(
      attn, wpbT, out, bproj, (int)BT, D_MODEL, D_MODEL, 0);

  (void)in_sizes; (void)n_in; (void)out_size;
}
